// StaticCacheModel_19911468384784
// MI455X (gfx1250) — matrix operand order repaired and checked by reading, not compiled, not run on hardware
//
#include <hip/hip_runtime.h>
#include <hip/hip_bf16.h>
#include <stdint.h>

#define BB 4
#define HH 32
#define SNEW 2048
#define DHEAD 128
#define MAXLEN 4096
#define KBLK 64

typedef __bf16 bf16x16 __attribute__((ext_vector_type(16)));
typedef __bf16 bf16x2 __attribute__((ext_vector_type(2)));
typedef float f32x8 __attribute__((ext_vector_type(8)));
typedef uint32_t u32x8 __attribute__((ext_vector_type(8)));

// Native fp32 -> bf16 conversion: clang fptrunc. If gfx1250 has
// v_cvt_pk_bf16_f32 the pair version lowers to one instruction; otherwise the
// backend emits its own RNE sequence (no worse than a manual one).
__device__ __forceinline__ uint32_t f2bf_pack(float lo, float hi) {
    bf16x2 t;
    t[0] = (__bf16)lo;
    t[1] = (__bf16)hi;
    return __builtin_bit_cast(uint32_t, t);
}
__device__ __forceinline__ uint16_t f2bf(float f) {
    return __builtin_bit_cast(uint16_t, (__bf16)f);
}

// XOR-butterfly max across the 16-lane half (ds_swizzle group-of-32 XOR mode)
__device__ __forceinline__ float rowmax16(float x) {
    x = fmaxf(x, __uint_as_float((uint32_t)__builtin_amdgcn_ds_swizzle(
                     (int)__float_as_uint(x), 0x041F)));
    x = fmaxf(x, __uint_as_float((uint32_t)__builtin_amdgcn_ds_swizzle(
                     (int)__float_as_uint(x), 0x081F)));
    x = fmaxf(x, __uint_as_float((uint32_t)__builtin_amdgcn_ds_swizzle(
                     (int)__float_as_uint(x), 0x101F)));
    x = fmaxf(x, __uint_as_float((uint32_t)__builtin_amdgcn_ds_swizzle(
                     (int)__float_as_uint(x), 0x201F)));
    return x;
}

// ---------------------------------------------------------------------------
// Flash attention: one block = 4 waves x 16 q-rows = 64 q rows; 64-key blocks.
// Keys sourced from (k | key_cache) per splice semantics; causal tril mask.
// ---------------------------------------------------------------------------
__global__ __launch_bounds__(128)
void attn_fa_kernel(const float* __restrict__ q, const float* __restrict__ k,
                    const float* __restrict__ v, const float* __restrict__ kc,
                    const float* __restrict__ vc, const int* __restrict__ sp,
                    const int* __restrict__ ep, float* __restrict__ out) {
    __shared__ uint16_t Ksh[KBLK * DHEAD];     // 64 keys x 128 dims, bf16, row-major
    __shared__ uint16_t VTsh[DHEAD * KBLK];    // transposed: [d][key], bf16
    __shared__ uint32_t Psh[4][16 * 32];       // per-wave P tile, 16 rows x 32 uints

    const int start = sp[0];
    const int end = ep[0];
    const int bh = blockIdx.y;                 // b*H + h
    const int i0 = blockIdx.x * 64;
    const int tid = threadIdx.x;
    const int w = tid >> 5;
    const int lane = tid & 31;
    const int ln = lane & 15;
    const int hi = lane >> 4;
    const int iw = i0 + w * 16;

    // 1/sqrt(128) * log2(e): softmax in base-2 domain
    const float scale2 = 0.08838834764831845f * 1.4426950408889634f;

    // all-ones bf16 B fragment for row-sum-by-WMMA
    u32x8 onesB;
#pragma unroll
    for (int i = 0; i < 8; ++i) onesB[i] = 0x3F803F80u;

    // ---- Load Q A-fragments (4 chunks of K=32 over head dim), convert to bf16
    u32x8 qa[4];
    {
        const float* qrow = q + ((size_t)bh * SNEW + (size_t)(iw + ln)) * DHEAD;
        const int kb = hi * 16;
#pragma unroll
        for (int kk = 0; kk < 4; ++kk) {
            const float* p0 = qrow + 32 * kk + kb;
            const float* p1 = qrow + 32 * kk + 8 + kb;
#pragma unroll
            for (int i = 0; i < 4; ++i) {
                qa[kk][i]     = f2bf_pack(p0[2 * i], p0[2 * i + 1]);
                qa[kk][4 + i] = f2bf_pack(p1[2 * i], p1[2 * i + 1]);
            }
        }
    }

    f32x8 o[8];
#pragma unroll
    for (int n = 0; n < 8; ++n) o[n] = (f32x8)0.0f;
    float m[8], l[8];
#pragma unroll
    for (int r = 0; r < 8; ++r) { m[r] = -1e20f; l[r] = 0.0f; }

    const int jhi = (end < i0 + 64) ? end : (i0 + 64);
    const int nblk = (jhi + KBLK - 1) / KBLK;

    for (int blk = 0; blk < nblk; ++blk) {
        const int j0 = blk * KBLK;
        __syncthreads();
        // ---- Stage K and V (64 keys x 128 dims) into LDS as bf16 (V transposed)
#pragma unroll
        for (int it = 0; it < 16; ++it) {
            int flat4 = tid + 128 * it;            // 0..2047 float4 units
            int row = flat4 >> 5;                  // key row in tile (0..63)
            int d = (flat4 & 31) * 4;              // head-dim base
            int t = j0 + row;
            bool innew = (t >= start) && (t < end);
            size_t idx_new = ((size_t)bh * SNEW + (size_t)(t - start)) * DHEAD + d;
            size_t idx_old = ((size_t)bh * MAXLEN + (size_t)t) * DHEAD + d;
            float4 kv = innew ? ((const float4*)(k + idx_new))[0]
                              : ((const float4*)(kc + idx_old))[0];
            float4 vv = innew ? ((const float4*)(v + idx_new))[0]
                              : ((const float4*)(vc + idx_old))[0];
            uint32_t* kd = (uint32_t*)&Ksh[(size_t)flat4 * 4];
            kd[0] = f2bf_pack(kv.x, kv.y);
            kd[1] = f2bf_pack(kv.z, kv.w);
            VTsh[(d + 0) * KBLK + row] = f2bf(vv.x);
            VTsh[(d + 1) * KBLK + row] = f2bf(vv.y);
            VTsh[(d + 2) * KBLK + row] = f2bf(vv.z);
            VTsh[(d + 3) * KBLK + row] = f2bf(vv.w);
        }
        __syncthreads();

        if (j0 < iw + 16) {  // wave-uniform: this wave has unmasked work here
            // ---- S = Q*K^T, four INTERLEAVED 16-col tiles (tile t covers keys
            //      j0 + 4c + t), 4 WMMAs each (K=32 chunks of D=128)
            f32x8 s[4];
#pragma unroll
            for (int t2 = 0; t2 < 4; ++t2) {
                s[t2] = (f32x8)0.0f;
                const int keyrow = 4 * ln + t2;
#pragma unroll
                for (int kk = 0; kk < 4; ++kk) {
                    const uint32_t* bp =
                        (const uint32_t*)&Ksh[keyrow * DHEAD + 32 * kk + hi * 16];
                    u32x8 bu;
#pragma unroll
                    for (int i = 0; i < 8; ++i) bu[i] = bp[i];
                    s[t2] = __builtin_amdgcn_wmma_f32_16x16x32_bf16(
                        false, __builtin_bit_cast(bf16x16, qa[kk]),
                        false, __builtin_bit_cast(bf16x16, bu),
                        (short)0, s[t2], false, false);
                }
            }

            // ---- Online softmax (base-2). C-layout: reg r = row (r + 8*hi),
            //      lane column c: tile t -> key j0 + 4c + t.
            const bool needmask = (j0 + KBLK - 1 > iw) || (j0 + KBLK > end);
            float alpha[8];
#pragma unroll
            for (int r = 0; r < 8; ++r) {
                const int irow = iw + r + 8 * hi;
                float t0 = s[0][r] * scale2;
                float t1 = s[1][r] * scale2;
                float t2v = s[2][r] * scale2;
                float t3 = s[3][r] * scale2;
                if (needmask) {
                    int jg = j0 + 4 * ln;
                    if (jg + 0 > irow || jg + 0 >= end) t0 = -1e30f;
                    if (jg + 1 > irow || jg + 1 >= end) t1 = -1e30f;
                    if (jg + 2 > irow || jg + 2 >= end) t2v = -1e30f;
                    if (jg + 3 > irow || jg + 3 >= end) t3 = -1e30f;
                }
                float mx = fmaxf(fmaxf(t0, t1), fmaxf(t2v, t3));
                float mnew = fmaxf(m[r], rowmax16(mx));
                alpha[r] = __builtin_amdgcn_exp2f(m[r] - mnew);
                float p0 = __builtin_amdgcn_exp2f(t0 - mnew);
                float p1 = __builtin_amdgcn_exp2f(t1 - mnew);
                float p2 = __builtin_amdgcn_exp2f(t2v - mnew);
                float p3 = __builtin_amdgcn_exp2f(t3 - mnew);
                // keys 4*ln .. 4*ln+3: two adjacent packed b32 stores
                uint32_t* pd = &Psh[w][(r + 8 * hi) * 32 + 2 * ln];
                pd[0] = f2bf_pack(p0, p1);
                pd[1] = f2bf_pack(p2, p3);
                m[r] = mnew;
#pragma unroll
                for (int n = 0; n < 8; ++n) o[n][r] *= alpha[r];
            }

            // ---- Re-read P (16x64) in A-matrix layout: two K=32 fragments
            u32x8 pa[2];
            {
                const uint32_t* pb = &Psh[w][ln * 32];
#pragma unroll
                for (int c = 0; c < 2; ++c) {
#pragma unroll
                    for (int i = 0; i < 4; ++i) {
                        pa[c][i]     = pb[c * 16 + hi * 8 + i];
                        pa[c][4 + i] = pb[c * 16 + hi * 8 + 4 + i];
                    }
                }
            }

            // ---- Row-sum of P via WMMA against all-ones B (chained K=64)
            f32x8 ssum = (f32x8)0.0f;
            ssum = __builtin_amdgcn_wmma_f32_16x16x32_bf16(
                false, __builtin_bit_cast(bf16x16, pa[0]),
                false, __builtin_bit_cast(bf16x16, onesB),
                (short)0, ssum, false, false);
            ssum = __builtin_amdgcn_wmma_f32_16x16x32_bf16(
                false, __builtin_bit_cast(bf16x16, pa[1]),
                false, __builtin_bit_cast(bf16x16, onesB),
                (short)0, ssum, false, false);
#pragma unroll
            for (int r = 0; r < 8; ++r) l[r] = l[r] * alpha[r] + ssum[r];

            // ---- O += P * V : 8 head-dim chunks x 2 chained WMMAs (K=64)
#pragma unroll
            for (int n = 0; n < 8; ++n) {
#pragma unroll
                for (int c = 0; c < 2; ++c) {
                    const uint32_t* vp = (const uint32_t*)
                        &VTsh[(n * 16 + ln) * KBLK + c * 32 + hi * 16];
                    u32x8 vb;
#pragma unroll
                    for (int i = 0; i < 8; ++i) vb[i] = vp[i];
                    o[n] = __builtin_amdgcn_wmma_f32_16x16x32_bf16(
                        false, __builtin_bit_cast(bf16x16, pa[c]),
                        false, __builtin_bit_cast(bf16x16, vb),
                        (short)0, o[n], false, false);
                }
            }
        }
    }

    // ---- Epilogue: O / l
#pragma unroll
    for (int r = 0; r < 8; ++r) {
        float inv = 1.0f / fmaxf(l[r], 1e-30f);
        int irow = iw + r + 8 * hi;
        float* orow = out + ((size_t)bh * SNEW + (size_t)irow) * DHEAD;
#pragma unroll
        for (int n = 0; n < 8; ++n) orow[n * 16 + ln] = o[n][r] * inv;
    }
}

// ---------------------------------------------------------------------------
// Cache splice: new_cache = concat(cache[:start], new, cache[end:]) per (b,h)
// Pure bandwidth; float4 vectorized. idx < N4 -> key cache, else value cache.
// ---------------------------------------------------------------------------
__global__ __launch_bounds__(256)
void splice_kernel(const float4* __restrict__ k4, const float4* __restrict__ v4,
                   const float4* __restrict__ kc4, const float4* __restrict__ vc4,
                   const int* __restrict__ sp, const int* __restrict__ ep,
                   float4* __restrict__ out4) {
    const size_t N4 = (size_t)BB * HH * MAXLEN * (DHEAD / 4);
    size_t idx = (size_t)blockIdx.x * blockDim.x + threadIdx.x;
    if (idx >= 2 * N4) return;
    const int start = sp[0];
    const int end = ep[0];
    const bool isv = idx >= N4;
    const size_t i = isv ? idx - N4 : idx;
    const int d4 = (int)(i % (DHEAD / 4));
    const size_t tmp = i / (DHEAD / 4);
    const int t = (int)(tmp % MAXLEN);
    const size_t bh = tmp / MAXLEN;
    float4 val;
    if (t >= start && t < end) {
        size_t src = (bh * SNEW + (size_t)(t - start)) * (DHEAD / 4) + d4;
        val = isv ? v4[src] : k4[src];
    } else {
        val = isv ? vc4[i] : kc4[i];
    }
    out4[idx] = val;
}

extern "C" void kernel_launch(void* const* d_in, const int* in_sizes, int n_in,
                              void* d_out, int out_size, void* d_ws, size_t ws_size,
                              hipStream_t stream) {
    const float* q  = (const float*)d_in[0];
    const float* k  = (const float*)d_in[1];
    const float* v  = (const float*)d_in[2];
    const float* kc = (const float*)d_in[3];
    const float* vc = (const float*)d_in[4];
    const int* sp   = (const int*)d_in[5];
    const int* ep   = (const int*)d_in[6];
    float* out = (float*)d_out;

    // Output layout: [attn out | new_key_cache | new_value_cache]
    const size_t attn_elems = (size_t)BB * HH * SNEW * DHEAD;

    dim3 gridA(SNEW / 64, BB * HH);
    attn_fa_kernel<<<gridA, 128, 0, stream>>>(q, k, v, kc, vc, sp, ep, out);

    const size_t N4total = 2ull * BB * HH * MAXLEN * (DHEAD / 4);
    const unsigned blocks = (unsigned)((N4total + 255) / 256);
    splice_kernel<<<dim3(blocks), 256, 0, stream>>>(
        (const float4*)k, (const float4*)v, (const float4*)kc, (const float4*)vc,
        sp, ep, (float4*)(out + attn_elems));
}
